// MixtureOfAttention_52956946759873
// MI455X (gfx1250) — compile-verified, matrix operand order checked by reading, not run on hardware
//
#include <hip/hip_runtime.h>
#include <hip/hip_bf16.h>
#include <cstdint>

// ---------------- problem constants (fixed by setup_inputs) ----------------
constexpr int B_ = 2, N_ = 4096, D_ = 1024, G_ = 2, H_ = 16, DH_ = 64;
constexpr int NQ_ = 1024, NKV_ = 1024;
constexpr int DI_ = H_ * DH_;          // 1024
constexpr int KVP_ = 1056;             // 1025 keys padded to multiple of 32
constexpr int BGH_ = B_ * G_ * H_;     // 64
constexpr int KCD_ = 1152;             // int(NQ * 9/8)

typedef __bf16 bf16_t;
typedef __attribute__((ext_vector_type(16))) __bf16 v16bf;
typedef __attribute__((ext_vector_type(8)))  float  v8f;
typedef __attribute__((ext_vector_type(4)))  int    v4i_;

// ---------------- CDNA5 async global->LDS copy (ASYNCcnt-tracked) ----------------
#if defined(__has_builtin)
#  if __has_builtin(__builtin_amdgcn_global_load_async_to_lds_b128)
#    define MOA_ASYNC_LDS 1
#  endif
#  if __has_builtin(__builtin_amdgcn_s_wait_asynccnt)
#    define MOA_WAIT_ASYNC_BUILTIN 1
#  endif
#endif

__device__ __forceinline__ void async_copy16(bf16_t* lds, const bf16_t* g) {
#ifdef MOA_ASYNC_LDS
  __builtin_amdgcn_global_load_async_to_lds_b128(
      (__attribute__((address_space(1))) v4i_*)g,
      (__attribute__((address_space(3))) v4i_*)lds, 0, 0);
#else
  *(uint4*)lds = *(const uint4*)g;
#endif
}

template <int Nw>
__device__ __forceinline__ void wait_async() {
#ifdef MOA_ASYNC_LDS
#  ifdef MOA_WAIT_ASYNC_BUILTIN
  __builtin_amdgcn_s_wait_asynccnt(Nw);
#  else
  asm volatile("s_wait_asynccnt %0" ::"n"(Nw) : "memory");
#  endif
#endif
}

union FragU { v16bf v; uint4 q[2]; };

// A/B fragment loader for "rows x K" row-major bf16 source (global or LDS).
// lane<16: row=lane, K runs {k0..k0+7, k0+16..k0+23}
// lane>=16: row=lane-16, K runs {k0+8..k0+15, k0+24..k0+31}
__device__ __forceinline__ v16bf load_frag(const bf16_t* p, int row0, int ld,
                                           int k0, int lane) {
  int half = lane >> 4;
  int r = lane & 15;
  const bf16_t* base = p + (size_t)(row0 + r) * ld + k0 + half * 8;
  FragU u;
  u.q[0] = *(const uint4*)(base);
  u.q[1] = *(const uint4*)(base + 16);
  return u.v;
}

__device__ __forceinline__ v8f wmma_bf16(v16bf a, v16bf b, v8f c) {
  return __builtin_amdgcn_wmma_f32_16x16x32_bf16(false, a, false, b,
                                                 (short)0, c, false, false);
}

// ---------------- block reductions (256 threads, 8 waves) ----------------
__device__ __forceinline__ float block_sum256(float v, float* red) {
  for (int o = 16; o; o >>= 1) v += __shfl_xor(v, o, 32);
  __syncthreads();
  if ((threadIdx.x & 31) == 0) red[threadIdx.x >> 5] = v;
  __syncthreads();
  float r = red[0];
  for (int i = 1; i < 8; ++i) r += red[i];
  return r;
}
__device__ __forceinline__ float block_max256(float v, float* red) {
  for (int o = 16; o; o >>= 1) v = fmaxf(v, __shfl_xor(v, o, 32));
  __syncthreads();
  if ((threadIdx.x & 31) == 0) red[threadIdx.x >> 5] = v;
  __syncthreads();
  float r = red[0];
  for (int i = 1; i < 8; ++i) r = fmaxf(r, red[i]);
  return r;
}

// ---------------- 0: zero output accumulator + counts ----------------
__global__ void moa_zero(float* out, float* counts) {
  size_t tot = (size_t)B_ * N_ * D_;
  size_t stride = (size_t)gridDim.x * blockDim.x;
  for (size_t e = (size_t)blockIdx.x * blockDim.x + threadIdx.x; e < tot; e += stride)
    out[e] = 0.0f;
  for (size_t e = (size_t)blockIdx.x * blockDim.x + threadIdx.x; e < (size_t)B_ * N_; e += stride)
    counts[e] = 0.0f;
}

// ---------------- 1: routing similarities sim[b,g,n] = x[b,n]·rt[g] ----------------
__global__ __launch_bounds__(256) void moa_sim(const float* __restrict__ x,
                                               const float* __restrict__ rtq,
                                               const float* __restrict__ rtkv,
                                               float* simq, float* simkv) {
  int wave = threadIdx.x >> 5, lane = threadIdx.x & 31;
  int out = blockIdx.x * 8 + wave;          // 0 .. 2*B*G*N-1
  int n = out % N_;
  int g = (out / N_) % G_;
  int b = (out / (N_ * G_)) % B_;
  int which = out / (N_ * G_ * B_);
  const float* rt = (which ? rtkv : rtq) + (size_t)g * D_;
  const float* xr = x + ((size_t)b * N_ + n) * D_;
  float acc = 0.0f;
  for (int d = lane * 4; d < D_; d += 128) {
    float4 xv = *(const float4*)(xr + d);
    float4 rv = *(const float4*)(rt + d);
    acc += xv.x * rv.x + xv.y * rv.y + xv.z * rv.z + xv.w * rv.w;
  }
  for (int o = 16; o; o >>= 1) acc += __shfl_xor(acc, o, 32);
  if (lane == 0) (which ? simkv : simq)[(size_t)(b * G_ + g) * N_ + n] = acc;
}

// ---------------- 2: coordinate descent (one block per (which,b,g) row) ----------------
__global__ __launch_bounds__(256) void moa_coor(const float* simq, const float* simkv,
                                                float* scq, float* sckv) {
  __shared__ float red[8];
  int row = blockIdx.x;            // 0..7
  int which = row >> 2;
  int bg = row & 3;
  const float* s_in = (which ? simkv : simq) + (size_t)bg * N_;
  float* s_out = (which ? sckv : scq) + (size_t)bg * N_;
  int tid = threadIdx.x;
  float sv[16], bv[16];
  for (int t = 0; t < 16; ++t) { sv[t] = s_in[tid + t * 256]; bv[t] = -sv[t]; }
  const float logk = __logf((float)KCD_);
  float ce = 4.0f, a = 0.0f;
  for (int it = 0; it < 20; ++it) {
    float ceff = fmaxf(ce, 0.03f);
    float lm = -3.4e38f;
    for (int t = 0; t < 16; ++t) lm = fmaxf(lm, sv[t] + bv[t]);
    float M = block_max256(lm, red);
    float ls = 0.0f;
    for (int t = 0; t < 16; ++t) ls += __expf((sv[t] + bv[t] - M) / ceff);
    float S = block_sum256(ls, red);
    a = ceff * (logk - __logf(S)) - M;       // ce*(logk - logsumexp((s+b)/ce))
    for (int t = 0; t < 16; ++t) bv[t] = -fmaxf(sv[t] + a, 0.0f);
    ce *= 0.7f;
  }
  const float inv_eps = 1.0f / 0.03f;
  for (int t = 0; t < 16; ++t)
    s_out[tid + t * 256] = __expf((sv[t] + a + bv[t]) * inv_eps);
}

// ---------------- 3: exact top-k by rank counting (matches lax.top_k tie-break) ----------------
__global__ __launch_bounds__(256) void moa_topk(const float* scq, const float* sckv,
                                                int* idxq, int* idxkv) {
  __shared__ float sc[N_];
  int row = blockIdx.x;
  int which = row >> 2;
  int bg = row & 3;
  const float* s_in = (which ? sckv : scq) + (size_t)bg * N_;
  int* out = (which ? idxkv : idxq) + (size_t)bg * NQ_;
  for (int i = threadIdx.x; i < N_; i += 256) sc[i] = s_in[i];
  __syncthreads();
  for (int t = 0; t < 16; ++t) {
    int i = threadIdx.x * 16 + t;
    float si = sc[i];
    int rank = 0;
    for (int j = 0; j < N_; ++j) {
      float sj = sc[j];
      rank += (sj > si) || (sj == si && j < i);
    }
    if (rank < NQ_) out[rank] = i;
  }
}

// ---------------- 4: gather routed tokens + rmsnorm + cast bf16 ----------------
__global__ __launch_bounds__(256) void moa_gather_norm(const float* __restrict__ x,
                                                       const int* idxq, const int* idxkv,
                                                       const float* gq, const float* gc,
                                                       bf16_t* qn, bf16_t* cn) {
  __shared__ float red[8];
  int which = blockIdx.y;
  int flat = blockIdx.x;
  int i = flat % NQ_;
  int g = (flat / NQ_) % G_;
  int b = flat / (NQ_ * G_);
  const int* idx = which ? idxkv : idxq;
  int token = idx[(size_t)(b * G_ + g) * NQ_ + i];
  const float* src = x + ((size_t)b * N_ + token) * D_;
  float ss = 0.0f;
  for (int d = threadIdx.x; d < D_; d += 256) { float v = src[d]; ss += v * v; }
  ss = block_sum256(ss, red);
  float scale = 32.0f / fmaxf(sqrtf(ss), 1e-12f);   // sqrt(D)=32
  const float* gamma = (which ? gc : gq) + (size_t)g * D_;
  bf16_t* dst = (which ? cn : qn) + ((size_t)(b * G_ + g) * NQ_ + i) * D_;
  for (int d = threadIdx.x; d < D_; d += 256)
    dst[d] = (bf16_t)(src[d] * scale * gamma[d]);
}

// ---------------- 5: cast weights fp32 -> bf16 ----------------
__global__ void moa_cast_w(const float* wq, const float* wkv, const float* wo,
                           bf16_t* wqb, bf16_t* wkvb, bf16_t* wob) {
  constexpr size_t NWQ = (size_t)G_ * DI_ * D_;       // 2,097,152
  constexpr size_t NWKV = (size_t)G_ * 2 * DI_ * D_;  // 4,194,304
  constexpr size_t NWO = (size_t)G_ * D_ * DI_;       // 2,097,152
  size_t tot = NWQ + NWKV + NWO;
  size_t stride = (size_t)gridDim.x * blockDim.x;
  for (size_t e = (size_t)blockIdx.x * blockDim.x + threadIdx.x; e < tot; e += stride) {
    if (e < NWQ) wqb[e] = (bf16_t)wq[e];
    else if (e < NWQ + NWKV) wkvb[e - NWQ] = (bf16_t)wkv[e - NWQ];
    else wob[e - NWQ - NWKV] = (bf16_t)wo[e - NWQ - NWKV];
  }
}

// ---------------- 6: batched bf16 WMMA GEMM  C[m,n] = sum_k A[m,k]*B[n,k] ----------------
// grid: (N/64, M/128, nbatch); block 256. Shared 64xK B tile double-buffered in LDS
// via CDNA5 async global->LDS copies (ASYNCcnt); A fragments direct from global.
__global__ __launch_bounds__(256) void moa_gemm_nt(const bf16_t* __restrict__ A,
                                                   const bf16_t* __restrict__ Bm,
                                                   bf16_t* __restrict__ Cb,
                                                   float* __restrict__ Cf,
                                                   int Ncols, int K,
                                                   long sA, long sB, long sC, int gmod) {
  __shared__ bf16_t Bs[2][64 * 32];   // 2 x 4 KB double buffer
  int z = blockIdx.z;
  const bf16_t* Az = A + (size_t)z * sA;
  const bf16_t* Bz = Bm + (size_t)(z % gmod) * sB;
  int wave = threadIdx.x >> 5, lane = threadIdx.x & 31;
  int m0 = blockIdx.y * 128 + wave * 16;
  int n0 = blockIdx.x * 64;
  // each thread owns one 16-byte segment of the 64x32 B tile
  int crow = threadIdx.x >> 2, cseg = threadIdx.x & 3;
  const bf16_t* gB = Bz + (size_t)(n0 + crow) * K + cseg * 8;
  bf16_t* lB0 = &Bs[0][crow * 32 + cseg * 8];
  bf16_t* lB1 = &Bs[1][crow * 32 + cseg * 8];

  async_copy16(lB0, gB);                       // prologue: tile 0
  v8f c0 = {}, c1 = {}, c2 = {}, c3 = {};
  int nsteps = K / 32;
  for (int i = 0; i < nsteps; ++i) {
    int buf = i & 1;
    if (i + 1 < nsteps) {
      async_copy16(buf ? lB0 : lB1, gB + (i + 1) * 32);  // next tile
      wait_async<1>();                                   // tile i landed
    } else {
      wait_async<0>();
    }
    __syncthreads();
    int k0 = i * 32;
    const bf16_t* Bt = &Bs[buf][0];
    v16bf a = load_frag(Az, m0, K, k0, lane);
    v16bf b0 = load_frag(Bt, 0, 32, 0, lane);
    v16bf b1 = load_frag(Bt, 16, 32, 0, lane);
    v16bf b2 = load_frag(Bt, 32, 32, 0, lane);
    v16bf b3 = load_frag(Bt, 48, 32, 0, lane);
    c0 = wmma_bf16(a, b0, c0);
    c1 = wmma_bf16(a, b1, c1);
    c2 = wmma_bf16(a, b2, c2);
    c3 = wmma_bf16(a, b3, c3);
    __builtin_prefetch(Az + (size_t)m0 * K + k0 + 64, 0, 1);  // global_prefetch_b8
    __syncthreads();   // everyone done reading buf before it is overwritten
  }
  int col = lane & 15;
  int rbase = m0 + ((lane >> 4) << 3);
  if (Cf) {
    float* C = Cf + (size_t)z * sC;
    for (int r = 0; r < 8; ++r) {
      size_t off = (size_t)(rbase + r) * Ncols + n0 + col;
      C[off] = c0[r]; C[off + 16] = c1[r]; C[off + 32] = c2[r]; C[off + 48] = c3[r];
    }
  } else {
    bf16_t* C = Cb + (size_t)z * sC;
    for (int r = 0; r < 8; ++r) {
      size_t off = (size_t)(rbase + r) * Ncols + n0 + col;
      C[off] = (bf16_t)c0[r]; C[off + 16] = (bf16_t)c1[r];
      C[off + 32] = (bf16_t)c2[r]; C[off + 48] = (bf16_t)c3[r];
    }
  }
}

// ---------------- 7: reshape to heads, fold 1/sqrt(dh), build K / V^T with null kv ----------------
__global__ void moa_reshape(const bf16_t* __restrict__ qpb, const bf16_t* __restrict__ kvpb,
                            const float* __restrict__ nullkv,
                            bf16_t* Qh, bf16_t* Kall, bf16_t* VT) {
  constexpr size_t QH_TOT = (size_t)BGH_ * NQ_ * DH_;
  constexpr size_t KV_TOT = (size_t)BGH_ * KVP_ * DH_;
  size_t tot = QH_TOT + 2 * KV_TOT;
  size_t stride = (size_t)gridDim.x * blockDim.x;
  for (size_t e = (size_t)blockIdx.x * blockDim.x + threadIdx.x; e < tot; e += stride) {
    if (e < QH_TOT) {
      int d = (int)(e % DH_);
      size_t r = e / DH_;
      int q = (int)(r % NQ_);
      int bgh = (int)(r / NQ_);
      int b = bgh >> 5, g = (bgh >> 4) & 1, h = bgh & 15;
      float v = (float)qpb[((size_t)(b * G_ + g) * NQ_ + q) * DI_ + h * DH_ + d];
      Qh[e] = (bf16_t)(v * 0.125f);                 // q * dh^-0.5
    } else if (e < QH_TOT + KV_TOT) {
      size_t e2 = e - QH_TOT;
      int d = (int)(e2 % DH_);
      size_t r = e2 / DH_;
      int kv = (int)(r % KVP_);
      int bgh = (int)(r / KVP_);
      int b = bgh >> 5, g = (bgh >> 4) & 1, h = bgh & 15;
      float v;
      if (kv == 0)            v = nullkv[((size_t)(0 * G_ + g) * H_ + h) * DH_ + d];
      else if (kv <= NKV_)    v = (float)kvpb[((size_t)(b * G_ + g) * NKV_ + (kv - 1)) * (2 * DI_) + h * DH_ + d];
      else                    v = 0.0f;
      Kall[(size_t)bgh * KVP_ * DH_ + (size_t)kv * DH_ + d] = (bf16_t)v;
    } else {
      size_t e2 = e - QH_TOT - KV_TOT;
      int d = (int)(e2 % DH_);
      size_t r = e2 / DH_;
      int kv = (int)(r % KVP_);
      int bgh = (int)(r / KVP_);
      int b = bgh >> 5, g = (bgh >> 4) & 1, h = bgh & 15;
      float v;
      if (kv == 0)            v = nullkv[((size_t)(1 * G_ + g) * H_ + h) * DH_ + d];
      else if (kv <= NKV_)    v = (float)kvpb[((size_t)(b * G_ + g) * NKV_ + (kv - 1)) * (2 * DI_) + DI_ + h * DH_ + d];
      else                    v = 0.0f;   // v_scale == 1.0 in forward
      VT[(size_t)bgh * DH_ * KVP_ + (size_t)d * KVP_ + kv] = (bf16_t)v;
    }
  }
}

// ---------------- 8: attention per (b,g,h), 64-query blocks, full-score-block in LDS ----------------
__global__ __launch_bounds__(256) void moa_attn(const bf16_t* __restrict__ Qh,
                                                const bf16_t* __restrict__ Kall,
                                                const bf16_t* __restrict__ VT,
                                                bf16_t* __restrict__ obf) {
  extern __shared__ char smem[];
  float* Ssh = (float*)smem;                             // 64 x KVP_ fp32
  bf16_t* Qs = (bf16_t*)(smem + (size_t)64 * KVP_ * 4);  // 64 x 64 bf16
  int bgh = blockIdx.x, qblk = blockIdx.y;
  int wave = threadIdx.x >> 5, lane = threadIdx.x & 31;
  // stage Q block via async global->LDS copies
  const bf16_t* Qg = Qh + ((size_t)bgh * NQ_ + (size_t)qblk * 64) * DH_;
  for (int e = threadIdx.x * 8; e < 64 * 64; e += 256 * 8)
    async_copy16(Qs + e, Qg + e);
  wait_async<0>();
  __syncthreads();
  // S = Q K^T   (264 16x16 tiles over 8 waves)
  const bf16_t* Kg = Kall + (size_t)bgh * KVP_ * DH_;
  for (int t = wave; t < 4 * (KVP_ / 16); t += 8) {
    int m0 = (t & 3) * 16;
    int n0 = (t >> 2) * 16;
    v8f c = {};
    for (int k0 = 0; k0 < DH_; k0 += 32) {
      v16bf a = load_frag(Qs, m0, DH_, k0, lane);
      v16bf bb = load_frag(Kg, n0, DH_, k0, lane);
      c = wmma_bf16(a, bb, c);
    }
    int col = n0 + (lane & 15);
    int rbase = m0 + ((lane >> 4) << 3);
    for (int r = 0; r < 8; ++r) Ssh[(size_t)(rbase + r) * KVP_ + col] = c[r];
  }
  __syncthreads();
  // softmax over cols 0..1024 (incl. null col 0); write bf16 P in place
  for (int rr = 0; rr < 8; ++rr) {
    int row = wave * 8 + rr;
    float* Srow = Ssh + (size_t)row * KVP_;
    float vals[33];
    float m = -3.4e38f;
    for (int t = 0; t < 33; ++t) {
      int col = lane + t * 32;
      float v = (col <= NKV_) ? Srow[col] : -3.4e38f;
      vals[t] = v;
      m = fmaxf(m, v);
    }
    for (int o = 16; o; o >>= 1) m = fmaxf(m, __shfl_xor(m, o, 32));
    float s = 0.0f;
    for (int t = 0; t < 33; ++t)
      s += (vals[t] > -1e37f) ? __expf(vals[t] - m) : 0.0f;
    for (int o = 16; o; o >>= 1) s += __shfl_xor(s, o, 32);
    float inv = 1.0f / s;
    bf16_t* Prow = (bf16_t*)Srow;   // same row base, bf16 stride 2*KVP_
    for (int t = 0; t < 33; ++t) {
      int col = lane + t * 32;
      float p = (col <= NKV_) ? __expf(vals[t] - m) * inv : 0.0f;
      Prow[col] = (bf16_t)p;
    }
  }
  __syncthreads();
  // O = P V   (16 tiles over 8 waves), write heads-interleaved bf16 output
  const bf16_t* Vg = VT + (size_t)bgh * DH_ * KVP_;
  const bf16_t* Pb = (const bf16_t*)Ssh;
  int b = bgh >> 5, g = (bgh >> 4) & 1, h = bgh & 15;
  for (int t = wave; t < 16; t += 8) {
    int m0 = (t & 3) * 16;
    int n0 = (t >> 2) * 16;
    v8f c = {};
    for (int k0 = 0; k0 < KVP_; k0 += 32) {
      v16bf a = load_frag(Pb, m0, 2 * KVP_, k0, lane);
      v16bf bb = load_frag(Vg, n0, KVP_, k0, lane);
      c = wmma_bf16(a, bb, c);
    }
    int col = n0 + (lane & 15);
    int rbase = m0 + ((lane >> 4) << 3);
    for (int r = 0; r < 8; ++r) {
      int q = qblk * 64 + rbase + r;
      obf[((size_t)(b * G_ + g) * NQ_ + q) * DI_ + h * DH_ + col] = (bf16_t)c[r];
    }
  }
}

// ---------------- 9: scatter-add into output (+counts) ----------------
__global__ __launch_bounds__(256) void moa_scatter(const float* __restrict__ attnout,
                                                   const int* __restrict__ idxq,
                                                   float* out, float* counts) {
  int flat = blockIdx.x;
  int i = flat % NQ_;
  int g = (flat / NQ_) % G_;
  int b = flat / (NQ_ * G_);
  int token = idxq[(size_t)(b * G_ + g) * NQ_ + i];
  const float* src = attnout + ((size_t)(b * G_ + g) * NQ_ + i) * D_;
  float* dst = out + ((size_t)b * N_ + token) * D_;
  for (int d = threadIdx.x; d < D_; d += 256) atomicAdd(dst + d, src[d]);
  if (threadIdx.x == 0) atomicAdd(counts + (size_t)b * N_ + token, 1.0f);
}

// ---------------- 10: finalize (mean or null_token) ----------------
__global__ void moa_finalize(float* out, const float* counts, const float* null_token) {
  size_t tot = (size_t)B_ * N_ * D_;
  size_t stride = (size_t)gridDim.x * blockDim.x;
  for (size_t e = (size_t)blockIdx.x * blockDim.x + threadIdx.x; e < tot; e += stride) {
    int d = (int)(e % D_);
    size_t bn = e / D_;
    float c = counts[bn];
    out[e] = (c == 0.0f) ? null_token[d] : out[e] / fmaxf(c, 1e-5f);
  }
}

// ---------------- host launch ----------------
extern "C" void kernel_launch(void* const* d_in, const int* in_sizes, int n_in,
                              void* d_out, int out_size, void* d_ws, size_t ws_size,
                              hipStream_t stream) {
  const float* x       = (const float*)d_in[0];
  const float* rt_q    = (const float*)d_in[1];
  const float* rt_kv   = (const float*)d_in[2];
  const float* gamma_q = (const float*)d_in[3];
  const float* gamma_c = (const float*)d_in[4];
  const float* wq      = (const float*)d_in[5];
  const float* wkv     = (const float*)d_in[6];
  const float* wo      = (const float*)d_in[7];
  const float* null_kv = (const float*)d_in[8];
  const float* null_tk = (const float*)d_in[9];
  float* out = (float*)d_out;

  char* ws = (char*)d_ws;
  size_t off = 0;
  auto alloc = [&](size_t bytes) -> void* {
    void* p = ws + off;
    off = (off + bytes + 255) & ~(size_t)255;
    return p;
  };
  float* simq   = (float*)alloc((size_t)B_ * G_ * N_ * 4);
  float* simkv  = (float*)alloc((size_t)B_ * G_ * N_ * 4);
  float* scq    = (float*)alloc((size_t)B_ * G_ * N_ * 4);
  float* sckv   = (float*)alloc((size_t)B_ * G_ * N_ * 4);
  int*   idxq   = (int*)alloc((size_t)B_ * G_ * NQ_ * 4);
  int*   idxkv  = (int*)alloc((size_t)B_ * G_ * NQ_ * 4);
  float* counts = (float*)alloc((size_t)B_ * N_ * 4);
  bf16_t* qn    = (bf16_t*)alloc((size_t)B_ * G_ * NQ_ * D_ * 2);
  bf16_t* cn    = (bf16_t*)alloc((size_t)B_ * G_ * NKV_ * D_ * 2);
  bf16_t* wqb   = (bf16_t*)alloc((size_t)G_ * DI_ * D_ * 2);
  bf16_t* wkvb  = (bf16_t*)alloc((size_t)G_ * 2 * DI_ * D_ * 2);
  bf16_t* wob   = (bf16_t*)alloc((size_t)G_ * D_ * DI_ * 2);
  bf16_t* qpb   = (bf16_t*)alloc((size_t)B_ * G_ * NQ_ * DI_ * 2);
  bf16_t* kvpb  = (bf16_t*)alloc((size_t)B_ * G_ * NKV_ * 2 * DI_ * 2);
  bf16_t* Qh    = (bf16_t*)alloc((size_t)BGH_ * NQ_ * DH_ * 2);
  bf16_t* Kall  = (bf16_t*)alloc((size_t)BGH_ * KVP_ * DH_ * 2);
  bf16_t* VT    = (bf16_t*)alloc((size_t)BGH_ * DH_ * KVP_ * 2);
  bf16_t* obf   = (bf16_t*)alloc((size_t)B_ * G_ * NQ_ * DI_ * 2);
  float* attnout = (float*)alloc((size_t)B_ * G_ * NQ_ * D_ * 4);
  (void)ws_size; (void)in_sizes; (void)n_in; (void)out_size;

  // 0) zero accumulators
  moa_zero<<<2048, 256, 0, stream>>>(out, counts);
  // 1) routing similarities
  moa_sim<<<(2 * B_ * G_ * N_) / 8, 256, 0, stream>>>(x, rt_q, rt_kv, simq, simkv);
  // 2) coordinate descent
  moa_coor<<<2 * B_ * G_, 256, 0, stream>>>(simq, simkv, scq, sckv);
  // 3) top-k selection
  moa_topk<<<2 * B_ * G_, 256, 0, stream>>>(scq, sckv, idxq, idxkv);
  // 4) gather + rmsnorm + bf16 cast
  moa_gather_norm<<<dim3(B_ * G_ * NQ_, 2), 256, 0, stream>>>(x, idxq, idxkv,
                                                              gamma_q, gamma_c, qn, cn);
  // 5) weights -> bf16
  moa_cast_w<<<8192, 256, 0, stream>>>(wq, wkv, wo, wqb, wkvb, wob);
  // 6) q projection: (B*G) x [1024x1024] @ [1024x1024]^T
  moa_gemm_nt<<<dim3(DI_ / 64, NQ_ / 128, B_ * G_), 256, 0, stream>>>(
      qn, wqb, qpb, nullptr, DI_, D_,
      (long)NQ_ * D_, (long)DI_ * D_, (long)NQ_ * DI_, G_);
  // 7) kv projection: N = 2048
  moa_gemm_nt<<<dim3(2 * DI_ / 64, NKV_ / 128, B_ * G_), 256, 0, stream>>>(
      cn, wkvb, kvpb, nullptr, 2 * DI_, D_,
      (long)NKV_ * D_, (long)2 * DI_ * D_, (long)NKV_ * 2 * DI_, G_);
  // 8) reshape into heads + null kv + V transpose
  moa_reshape<<<8192, 256, 0, stream>>>(qpb, kvpb, null_kv, Qh, Kall, VT);
  // 9) attention
  size_t attn_smem = (size_t)64 * KVP_ * 4 + (size_t)64 * 64 * 2;  // 278528 B < 320 KB
  moa_attn<<<dim3(BGH_, NQ_ / 64), 256, attn_smem, stream>>>(Qh, Kall, VT, obf);
  // 10) output projection (fp32 out)
  moa_gemm_nt<<<dim3(D_ / 64, NQ_ / 128, B_ * G_), 256, 0, stream>>>(
      obf, wob, nullptr, attnout, D_, DI_,
      (long)NQ_ * DI_, (long)D_ * DI_, (long)NQ_ * D_, G_);
  // 11) scatter-add
  moa_scatter<<<B_ * G_ * NQ_, 256, 0, stream>>>(attnout, idxq, out, counts);
  // 12) finalize
  moa_finalize<<<8192, 256, 0, stream>>>(out, counts, null_tk);
}
